// MyViT_120259085169
// MI455X (gfx1250) — matrix operand order repaired and checked by reading, not compiled, not run on hardware
//
#include <hip/hip_runtime.h>
#include <hip/hip_bf16.h>
#include <cstdint>

// ---------------------------------------------------------------------------
// ViT-Base forward for MI455X (gfx1250): bf16 WMMA GEMMs with f32 accumulate.
// GEMM tile staging uses the CDNA5 Tensor Data Mover (TENSOR_LOAD_TO_LDS,
// TENSORcnt + s_wait_tensorcnt); double-buffered LDS, one barrier per K-step.
// Block tile 128x128 (or 128x64 for N=64 GEMMs), 4 wave32s, each wave owns a
// 64x64 (64x32) quadrant -> 16 (8) v_wmma_f32_16x16x32_bf16 per wave per step,
// 1 ds_load_b128 per WMMA.
// All GEMM K dims are multiples of 32 (token dim padded to SP=256); tile
// overreads land in ws slack / are zero-filled by TDM OOB handling and only
// feed store-guarded (discarded) outputs. B is always passed transposed [N][K].
// ---------------------------------------------------------------------------

typedef __attribute__((ext_vector_type(16))) __bf16 v16bf;
typedef __attribute__((ext_vector_type(8)))  float  v8f;

#define NB   32          // batch
#define SQ   197         // tokens
#define SP   256         // padded tokens (multiple of 32: clean WMMA K)
#define DIM  768
#define NH   12
#define DH   64
#define NL   12
#define MLPD 3072
#define NPP  196
#define ROWS (NB*SQ)     // 6304

#if __has_builtin(__builtin_amdgcn_tensor_load_to_lds) && __has_builtin(__builtin_amdgcn_s_wait_tensorcnt)
#define USE_TDM 1
#warning "CDNA5 TDM path enabled: tensor_load_to_lds staging"
#else
#define USE_TDM 0
#warning "TDM builtins unavailable: per-thread b128 staging fallback"
#endif

static __device__ __forceinline__ uint16_t f2bf(float f) {
  uint32_t u = __float_as_uint(f);
  uint32_t r = u + 0x7FFFu + ((u >> 16) & 1u);   // round-to-nearest-even
  return (uint16_t)(r >> 16);
}

#define BK 32
#define GEMM_THREADS 128

enum { EPI_NONE = 0, EPI_BIAS = 1, EPI_BIAS_GELU = 2 };
enum { OUT_F32 = 0, OUT_F32_ACC = 1, OUT_BF16 = 2 };

#if USE_TDM
// Issue one TDM 2-D tile load: tile_dim0=BK (K elems), tile_dim1=rows_tile of
// bf16 from a row-major tensor (row stride `ld` elems) into LDS, padding LDS
// by 16 B after every 64 B row => 80 B pitch. tensor_dim0/1 = remaining K /
// remaining rows (hardware zero-fills tile overreach).
// D# packing per CDNA5 ISA.
static __device__ __forceinline__ void tdm_tile_load(const uint16_t* gsrc,
                                                     uint32_t lds_byte_addr,
                                                     int rows_tile, int rows_rem,
                                                     int k_rem, int ld) {
  typedef __attribute__((ext_vector_type(4))) unsigned int v4u;
  typedef __attribute__((ext_vector_type(4))) int          v4i;
  typedef __attribute__((ext_vector_type(8))) int          v8i;
  const uint64_t ga = (uint64_t)(uintptr_t)gsrc;
  v4u g0;
  g0[0] = 1u;                                        // count=1 (valid user D#)
  g0[1] = lds_byte_addr;                             // lds_addr
  g0[2] = (uint32_t)ga;                              // global_addr[31:0]
  g0[3] = (uint32_t)((ga >> 32) & 0x01FFFFFFu) | (2u << 30);  // addr[56:32]|type=2
  const uint32_t td0 = (uint32_t)k_rem;              // tensor_dim0 (elems)
  const uint32_t td1 = (uint32_t)rows_rem;           // tensor_dim1 (rows)
  v8i g1;
  g1[0] = (int)((1u << 16)        // data_size = 1 -> 2-byte elements
              | (1u << 20)        // pad_enable
              | (3u << 22)        // pad_interval: 2^3*8 B = 64 B of data
              | (3u << 25));      // pad_amount: 4 dwords = 16 B
  g1[1] = (int)((td0 & 0xFFFFu) << 16);              // [47:32]=0, td0[15:0]
  g1[2] = (int)((td0 >> 16) | ((td1 & 0xFFFFu) << 16));
  g1[3] = (int)((td1 >> 16) | ((uint32_t)BK << 16)); // tile_dim0 = 32
  g1[4] = (int)(uint32_t)rows_tile;                  // tile_dim1; tile_dim2 = 0
  g1[5] = (int)(uint32_t)ld;                         // tensor_dim0_stride[31:0]
  g1[6] = 0;                                         // stride0[47:32], stride1[15:0]
  g1[7] = 0;
  v4i z4 = {0, 0, 0, 0};
#if defined(__clang_major__) && __clang_major__ >= 23
  v8i z8 = {0, 0, 0, 0, 0, 0, 0, 0};
  __builtin_amdgcn_tensor_load_to_lds(g0, g1, z4, z4, z8, 0);
#else
  __builtin_amdgcn_tensor_load_to_lds(g0, g1, z4, z4, 0);
#endif
}
#endif  // USE_TDM

// ---------------------------------------------------------------------------
// Batched GEMM: C[z] (+)= A[z] @ BT[z]^T (+bias) (gelu?)
// A:  [M,K] bf16 row-major (lda); BT: [N,K] bf16 row-major (ldbt)
// C:  [M,N] f32/bf16 (ldc). Batch: z1 = z % Z1, z2 = z / Z1.
// Tile TBM x TBN x 32; 4 wave32s in 2x2, each wave = (TBM/2)x(TBN/2).
// ---------------------------------------------------------------------------
template<int EPI, int OUTM, int TBM, int TBN>
__global__ void __launch_bounds__(GEMM_THREADS)
gemm_bt_k(const uint16_t* __restrict__ A,  long long sA1, long long sA2, int lda,
          const uint16_t* __restrict__ BT, long long sB1, long long sB2, int ldbt,
          const float* __restrict__ bias, long long sb1,
          void* __restrict__ Cvoid, long long sC1, long long sC2, int ldc,
          int M, int N, int K, int Z1)
{
  constexpr int RI = TBM / 32;       // 16-row fragments per wave
  constexpr int RJ = TBN / 32;       // 16-col fragments per wave
  __shared__ uint4 As4[2][TBM * 5];  // [buf][row*5 + chunk] : rows x 80 B
  __shared__ uint4 Bs4[2][TBN * 5];

  const int z  = blockIdx.z;
  const int z1 = z % Z1, z2 = z / Z1;
  A  += (long long)z1 * sA1 + (long long)z2 * sA2;
  BT += (long long)z1 * sB1 + (long long)z2 * sB2;
  const float* biasp = bias + (long long)z1 * sb1;
  float*    Cf = (float*)Cvoid    + (long long)z1 * sC1 + (long long)z2 * sC2;
  uint16_t* Cb = (uint16_t*)Cvoid + (long long)z1 * sC1 + (long long)z2 * sC2;

  const int m0  = blockIdx.y * TBM;
  const int n0  = blockIdx.x * TBN;
  const int tid  = threadIdx.x;
  const int lane = tid & 31;
  const int wid  = tid >> 5;
  const int rowbase = (wid >> 1) * (TBM / 2);   // wave quadrant origin
  const int colbase = (wid & 1)  * (TBN / 2);
  const int half = lane >> 4;        // ISA lane-group (0/1)
  const int l16  = lane & 15;

  v8f acc[RI][RJ];
  #pragma unroll
  for (int i = 0; i < RI; ++i)
    #pragma unroll
    for (int j = 0; j < RJ; ++j)
      #pragma unroll
      for (int r = 0; r < 8; ++r) acc[i][j][r] = 0.f;

  const int nsteps = K / BK;

#if USE_TDM
  const uint16_t* Abase = A  + (long long)m0 * lda;
  const uint16_t* Bbase = BT + (long long)n0 * ldbt;
  const int Mrem = M - m0 > 0 ? M - m0 : 1;
  const int Nrem = N - n0 > 0 ? N - n0 : 1;
  if (wid == 0) {                       // one wave drives the TDM
    tdm_tile_load(Abase, (uint32_t)(uintptr_t)&As4[0][0], TBM, Mrem, K, lda);
    tdm_tile_load(Bbase, (uint32_t)(uintptr_t)&Bs4[0][0], TBN, Nrem, K, ldbt);
    __builtin_amdgcn_s_wait_tensorcnt(0);
  }
  __syncthreads();
#endif

  for (int s = 0; s < nsteps; ++s) {
#if USE_TDM
    const int cur = s & 1;
    if (s + 1 < nsteps && wid == 0) {   // issue next tiles into other buffer
      const int k = (s + 1) * BK;
      tdm_tile_load(Abase + k, (uint32_t)(uintptr_t)&As4[cur ^ 1][0], TBM, Mrem, K - k, lda);
      tdm_tile_load(Bbase + k, (uint32_t)(uintptr_t)&Bs4[cur ^ 1][0], TBN, Nrem, K - k, ldbt);
    }
#else
    const int cur = 0;                  // synchronous single-buffer fallback
    if (s) __syncthreads();
    const int k0 = s * BK;
    #pragma unroll
    for (int it = 0; it < TBM / 32; ++it) {
      int c = it * GEMM_THREADS + tid; int row = c >> 2; int ch = c & 3;
      As4[0][row * 5 + ch] = *(const uint4*)(A + (long long)(m0 + row) * lda + k0 + ch * 8);
    }
    #pragma unroll
    for (int it = 0; it < TBN / 32; ++it) {
      int c = it * GEMM_THREADS + tid; int row = c >> 2; int ch = c & 3;
      Bs4[0][row * 5 + ch] = *(const uint4*)(BT + (long long)(n0 + row) * ldbt + k0 + ch * 8);
    }
    __syncthreads();
#endif

    // ---- fragments per ISA VGPR layouts (CDNA5 ISA §7.12.2) ----
    const uint32_t* Asu = (const uint32_t*)&As4[cur][0];
    const uint32_t* Bsu = (const uint32_t*)&Bs4[cur][0];
    union Frag { v16bf v; uint32_t u[8]; };
    Frag a[RI];
    #pragma unroll
    for (int i = 0; i < RI; ++i) {
      const uint32_t* ar = Asu + (rowbase + i * 16 + l16) * 20;
      #pragma unroll
      for (int t = 0; t < 4; ++t) {
        a[i].u[t]     = ar[half * 4 + t];
        a[i].u[4 + t] = ar[8 + half * 4 + t];
      }
    }
    #pragma unroll
    for (int j = 0; j < RJ; ++j) {      // one B fragment live at a time
      Frag b;
      const uint32_t* br = Bsu + (colbase + j * 16 + l16) * 20 + half * 4;
      #pragma unroll
      for (int q = 0; q < 8; ++q) b.u[q] = br[q < 4 ? q : q + 4];
      #pragma unroll
      for (int i = 0; i < RI; ++i)
        acc[i][j] = __builtin_amdgcn_wmma_f32_16x16x32_bf16(
            false, a[i].v, false, b.v, (short)0, acc[i][j], false, false);
    }

#if USE_TDM
    if (wid == 0) __builtin_amdgcn_s_wait_tensorcnt(0);
    __syncthreads();
#endif
  }

  // ---- epilogue: C/D layout = VGPR r -> M=r (+8 for lanes 16-31), N=l16 ----
  #pragma unroll
  for (int i = 0; i < RI; ++i) {
    #pragma unroll
    for (int j = 0; j < RJ; ++j) {
      int gn = n0 + colbase + j * 16 + l16;
      #pragma unroll
      for (int r = 0; r < 8; ++r) {
        int gm = m0 + rowbase + i * 16 + half * 8 + r;
        if (gm < M && gn < N) {
          float v = acc[i][j][r];
          if (EPI >= EPI_BIAS)      v += biasp[gn];
          if (EPI == EPI_BIAS_GELU) v = 0.5f * v * (1.f + erff(v * 0.70710678118654752f));
          long long off = (long long)gm * ldc + gn;
          if (OUTM == OUT_F32)          Cf[off] = v;
          else if (OUTM == OUT_F32_ACC) Cf[off] += v;
          else                          Cb[off] = f2bf(v);
        }
      }
    }
  }
}

// ---------------------------------------------------------------------------
// Support kernels
// ---------------------------------------------------------------------------
// transpose-convert: out[N][K] bf16 <- in[K][N] f32 (coalesced writes)
__global__ void cvtT_k(const float* __restrict__ in, uint16_t* __restrict__ o,
                       int Kd, int Nd) {
  long long idx = (long long)blockIdx.x * 256 + threadIdx.x;
  if (idx >= (long long)Kd * Nd) return;
  int k = (int)(idx % Kd); long long n = idx / Kd;
  o[idx] = f2bf(in[(long long)k * Nd + n]);
}

// batched 64x64 transpose-convert (per-head QKV weights): out[b][e][d]=in[b][d][e]
__global__ void cvtT64_k(const float* __restrict__ in, uint16_t* __restrict__ o, int nbat) {
  long long idx = (long long)blockIdx.x * 256 + threadIdx.x;
  if (idx >= (long long)nbat * 4096) return;
  int r = (int)(idx & 4095); long long b = idx >> 12;
  int e = r >> 6, d = r & 63;
  o[idx] = f2bf(in[b * 4096 + d * 64 + e]);
}

__global__ void im2col_k(const float* __restrict__ img, uint16_t* __restrict__ o) {
  long long idx = (long long)blockIdx.x * 256 + threadIdx.x;
  if (idx >= (long long)NB * NPP * DIM) return;
  int d = (int)(idx % DIM); long long t = idx / DIM;
  int p = (int)(t % NPP);   int n = (int)(t / NPP);
  int c  = d >> 8;          int py = (d >> 4) & 15; int px = d & 15;
  int pr = p / 14, pc = p % 14;
  float v = img[(((long long)n * 3 + c) * 224 + (pr * 16 + py)) * 224 + (pc * 16 + px)];
  o[idx] = f2bf(v);
}

__global__ void embed_finalize_k(float* __restrict__ x, const float* __restrict__ cls,
                                 const float* __restrict__ pos) {
  long long idx = (long long)blockIdx.x * 256 + threadIdx.x;
  if (idx >= (long long)ROWS * DIM) return;
  int d = (int)(idx % DIM); long long t = idx / DIM;
  int s = (int)(t % SQ);
  if (s == 0) x[idx] = cls[d] + pos[d];
  else        x[idx] += pos[(long long)s * DIM + d];
}

__global__ void __launch_bounds__(256)
layernorm_k(const float* __restrict__ x, const float* __restrict__ g,
            const float* __restrict__ b, uint16_t* __restrict__ o) {
  __shared__ float red[256];
  int row = blockIdx.x, tid = threadIdx.x;
  const float* xr = x + (long long)row * DIM;
  float v0 = xr[tid], v1 = xr[tid + 256], v2 = xr[tid + 512];
  red[tid] = v0 + v1 + v2; __syncthreads();
  for (int s = 128; s > 0; s >>= 1) { if (tid < s) red[tid] += red[tid + s]; __syncthreads(); }
  float mean = red[0] * (1.0f / DIM); __syncthreads();
  float d0 = v0 - mean, d1 = v1 - mean, d2 = v2 - mean;
  red[tid] = d0 * d0 + d1 * d1 + d2 * d2; __syncthreads();
  for (int s = 128; s > 0; s >>= 1) { if (tid < s) red[tid] += red[tid + s]; __syncthreads(); }
  float inv = rsqrtf(red[0] * (1.0f / DIM) + 1e-5f);
  uint16_t* orow = o + (long long)row * DIM;
  orow[tid]       = f2bf(d0 * inv * g[tid]       + b[tid]);
  orow[tid + 256] = f2bf(d1 * inv * g[tid + 256] + b[tid + 256]);
  orow[tid + 512] = f2bf(d2 * inv * g[tid + 512] + b[tid + 512]);
}

// vT[n,h,d,t] = v[n,t,h*64+d] (t<197 else 0): att@v takes vT as B-transposed.
__global__ void vT_build_k(const uint16_t* __restrict__ vbp, uint16_t* __restrict__ vT) {
  long long idx = (long long)blockIdx.x * 256 + threadIdx.x;
  if (idx >= (long long)NB * NH * DH * SP) return;
  int t = (int)(idx % SP); long long r = idx / SP;
  int d = (int)(r % DH);  r /= DH;
  int h = (int)(r % NH);  int n = (int)(r / NH);
  vT[idx] = (t < SQ) ? vbp[((long long)(n * SQ + t)) * DIM + h * DH + d] : (uint16_t)0;
}

// softmax over one score row (scale 1/sqrt(64) folded in); cols [197,256)=0
__global__ void __launch_bounds__(256)
softmax_k(const float* __restrict__ sc, uint16_t* __restrict__ att) {
  __shared__ float red[256];
  int row = blockIdx.x, tid = threadIdx.x;       // row in [0, NB*NH*SQ)
  long long base = (long long)row * SP;
  const float* sr = sc + base;
  uint16_t* ar = att + base;
  float v = (tid < SQ) ? sr[tid] * 0.125f : -3.0e38f;
  red[tid] = v; __syncthreads();
  for (int s = 128; s > 0; s >>= 1) { if (tid < s) red[tid] = fmaxf(red[tid], red[tid + s]); __syncthreads(); }
  float mx = red[0]; __syncthreads();
  float e = (tid < SQ) ? __expf(v - mx) : 0.f;
  red[tid] = e; __syncthreads();
  for (int s = 128; s > 0; s >>= 1) { if (tid < s) red[tid] += red[tid + s]; __syncthreads(); }
  float inv = 1.f / red[0];
  ar[tid] = (tid < SQ) ? f2bf(e * inv) : (uint16_t)0;
}

__global__ void __launch_bounds__(256)
head_k(const float* __restrict__ x, const float* __restrict__ Wo,
       const float* __restrict__ bo, float* __restrict__ out) {
  __shared__ float red[256];
  int n = blockIdx.x, tid = threadIdx.x;
  const float* xr = x + (long long)n * SQ * DIM;   // CLS row
  float s = xr[tid] * Wo[tid] + xr[tid + 256] * Wo[tid + 256] + xr[tid + 512] * Wo[tid + 512];
  red[tid] = s; __syncthreads();
  for (int t = 128; t > 0; t >>= 1) { if (tid < t) red[tid] += red[tid + t]; __syncthreads(); }
  if (tid == 0) out[n] = red[0] + bo[0];
}

// ---------------------------------------------------------------------------
// Host orchestration (needs ~250 MB workspace; everything rebuilt per call)
// ---------------------------------------------------------------------------
static inline size_t alignup(size_t x) { return (x + 255) & ~(size_t)255; }

extern "C" void kernel_launch(void* const* d_in, const int* in_sizes, int n_in,
                              void* d_out, int out_size, void* d_ws, size_t ws_size,
                              hipStream_t stream) {
  (void)in_sizes; (void)n_in; (void)out_size; (void)ws_size;
  const float* images = (const float*)d_in[0];
  const float* W_map  = (const float*)d_in[1];
  const float* b_map  = (const float*)d_in[2];
  const float* cls_t  = (const float*)d_in[3];
  const float* pos    = (const float*)d_in[4];
  const float* ln1g   = (const float*)d_in[5];
  const float* ln1b   = (const float*)d_in[6];
  const float* Wq     = (const float*)d_in[7];
  const float* bq     = (const float*)d_in[8];
  const float* Wk     = (const float*)d_in[9];
  const float* bk     = (const float*)d_in[10];
  const float* Wv     = (const float*)d_in[11];
  const float* bv     = (const float*)d_in[12];
  const float* ln2g   = (const float*)d_in[13];
  const float* ln2b   = (const float*)d_in[14];
  const float* W1     = (const float*)d_in[15];
  const float* b1     = (const float*)d_in[16];
  const float* W2     = (const float*)d_in[17];
  const float* b2     = (const float*)d_in[18];
  const float* W_out  = (const float*)d_in[19];
  const float* b_out  = (const float*)d_in[20];
  float* out = (float*)d_out;

  char* w = (char*)d_ws; size_t off = 0;
  auto carve = [&](size_t bytes) -> void* { void* p = w + off; off = alignup(off + bytes); return p; };
  float*    xf   = (float*)   carve((size_t)ROWS * DIM * 4);           // residual stream
  uint16_t* pat  = (uint16_t*)carve((size_t)NB * NPP * DIM * 2);       // im2col (bf16)
  uint16_t* hn   = (uint16_t*)carve((size_t)ROWS * DIM * 2);           // LN out (bf16)
  uint16_t* qb   = (uint16_t*)carve((size_t)ROWS * DIM * 2);
  uint16_t* kb   = (uint16_t*)carve((size_t)ROWS * DIM * 2);
  uint16_t* vb   = (uint16_t*)carve((size_t)ROWS * DIM * 2);
  uint16_t* vT   = (uint16_t*)carve((size_t)NB * NH * DH * SP * 2);
  float*    sc   = (float*)   carve((size_t)NB * NH * SQ * SP * 4);
  uint16_t* att  = (uint16_t*)carve((size_t)NB * NH * SQ * SP * 2);
  uint16_t* act  = (uint16_t*)carve((size_t)ROWS * MLPD * 2);
  uint16_t* wmT  = (uint16_t*)carve((size_t)DIM * DIM * 2);
  uint16_t* wqT  = (uint16_t*)carve((size_t)NL * NH * DH * DH * 2);
  uint16_t* wkT  = (uint16_t*)carve((size_t)NL * NH * DH * DH * 2);
  uint16_t* wvT  = (uint16_t*)carve((size_t)NL * NH * DH * DH * 2);
  uint16_t* w1T  = (uint16_t*)carve((size_t)MLPD * DIM * 2);           // per-layer reuse
  uint16_t* w2T  = (uint16_t*)carve((size_t)DIM * MLPD * 2);
  (void)carve((size_t)2 << 20);   // tail slack: unguarded tile overreads land here

  // --- weight transpose-conversions (small ones all-layers up front) ---
  cvtT_k<<<dim3((DIM * DIM + 255) / 256), 256, 0, stream>>>(W_map, wmT, DIM, DIM);
  {
    int nb = (NL * NH * 4096 + 255) / 256;
    cvtT64_k<<<dim3(nb), 256, 0, stream>>>(Wq, wqT, NL * NH);
    cvtT64_k<<<dim3(nb), 256, 0, stream>>>(Wk, wkT, NL * NH);
    cvtT64_k<<<dim3(nb), 256, 0, stream>>>(Wv, wvT, NL * NH);
  }

  // --- patch embedding ---
  im2col_k<<<dim3((NB * NPP * DIM + 255) / 256), 256, 0, stream>>>(images, pat);
  gemm_bt_k<EPI_BIAS, OUT_F32, 128, 128><<<dim3(DIM / 128, 2, NB), GEMM_THREADS, 0, stream>>>(
      pat, (long long)NPP * DIM, 0, DIM,
      wmT, 0, 0, DIM,
      b_map, 0,
      xf + DIM, (long long)SQ * DIM, 0, DIM,
      NPP, DIM, DIM, NB);
  embed_finalize_k<<<dim3((ROWS * DIM + 255) / 256), 256, 0, stream>>>(xf, cls_t, pos);

  const int MT = (ROWS + 127) / 128;   // 50 row tiles

  for (int l = 0; l < NL; ++l) {
    // LN1 -> hn (bf16)
    layernorm_k<<<dim3(ROWS), 256, 0, stream>>>(xf, ln1g + l * DIM, ln1b + l * DIM, hn);

    // per-head QKV: z = head; BT = per-head transposed weight [e][d]
    const long long wl = (long long)l * NH * DH * DH;
    gemm_bt_k<EPI_BIAS, OUT_BF16, 128, 64><<<dim3(1, MT, NH), GEMM_THREADS, 0, stream>>>(
        hn, DH, 0, DIM,  wqT + wl, (long long)DH * DH, 0, DH,
        bq + (long long)l * NH * DH, DH,
        qb, DH, 0, DIM,  ROWS, DH, DH, NH);
    gemm_bt_k<EPI_BIAS, OUT_BF16, 128, 64><<<dim3(1, MT, NH), GEMM_THREADS, 0, stream>>>(
        hn, DH, 0, DIM,  wkT + wl, (long long)DH * DH, 0, DH,
        bk + (long long)l * NH * DH, DH,
        kb, DH, 0, DIM,  ROWS, DH, DH, NH);
    gemm_bt_k<EPI_BIAS, OUT_BF16, 128, 64><<<dim3(1, MT, NH), GEMM_THREADS, 0, stream>>>(
        hn, DH, 0, DIM,  wvT + wl, (long long)DH * DH, 0, DH,
        bv + (long long)l * NH * DH, DH,
        vb, DH, 0, DIM,  ROWS, DH, DH, NH);

    // v -> vT [n,h,64,256]
    vT_build_k<<<dim3((NB * NH * DH * SP + 255) / 256), 256, 0, stream>>>(vb, vT);

    // scores[n,h] = q[n,:,h,:] @ k[n,:,h,:]^T : BT is just k itself (z1=h, z2=n)
    gemm_bt_k<EPI_NONE, OUT_F32, 128, 128><<<dim3(2, 2, NB * NH), GEMM_THREADS, 0, stream>>>(
        qb, DH, (long long)SQ * DIM, DIM,
        kb, DH, (long long)SQ * DIM, DIM,
        b_map /*unused*/, 0,
        sc, (long long)SQ * SP, (long long)NH * SQ * SP, SP,
        SQ, SQ, DH, NH);

    // softmax rows -> att (bf16, cols [197,256) zeroed => K=256 is safe below)
    softmax_k<<<dim3(NB * NH * SQ), 256, 0, stream>>>(sc, att);

    // x += att @ v  (BT = vT)
    gemm_bt_k<EPI_NONE, OUT_F32_ACC, 128, 64><<<dim3(1, 2, NB * NH), GEMM_THREADS, 0, stream>>>(
        att, (long long)SQ * SP, (long long)NH * SQ * SP, SP,
        vT, (long long)DH * SP, (long long)NH * DH * SP, SP,
        b_map /*unused*/, 0,
        xf, DH, (long long)SQ * DIM, DIM,
        SQ, DH, SP, NH);

    // LN2 -> hn (reused)
    layernorm_k<<<dim3(ROWS), 256, 0, stream>>>(xf, ln2g + l * DIM, ln2b + l * DIM, hn);

    // MLP up: act = gelu(hn @ W1 + b1); BT = W1^T [3072,768]
    cvtT_k<<<dim3((DIM * MLPD + 255) / 256), 256, 0, stream>>>(
        W1 + (size_t)l * DIM * MLPD, w1T, DIM, MLPD);
    gemm_bt_k<EPI_BIAS_GELU, OUT_BF16, 128, 128><<<dim3(MLPD / 128, MT, 1), GEMM_THREADS, 0, stream>>>(
        hn, 0, 0, DIM,  w1T, 0, 0, DIM,
        b1 + (long long)l * MLPD, 0,
        act, 0, 0, MLPD,  ROWS, MLPD, DIM, 1);

    // MLP down + residual: x += act @ W2 + b2; BT = W2^T [768,3072]
    cvtT_k<<<dim3((MLPD * DIM + 255) / 256), 256, 0, stream>>>(
        W2 + (size_t)l * MLPD * DIM, w2T, MLPD, DIM);
    gemm_bt_k<EPI_BIAS, OUT_F32_ACC, 128, 128><<<dim3(DIM / 128, MT, 1), GEMM_THREADS, 0, stream>>>(
        act, 0, 0, MLPD,  w2T, 0, 0, MLPD,
        b2 + (long long)l * DIM, 0,
        xf, 0, 0, DIM,  ROWS, DIM, MLPD, 1);
  }

  // head: out[n] = x[n,0,:] . W_out + b_out
  head_k<<<dim3(NB), 256, 0, stream>>>(xf, W_out, b_out, out);
}
